// CQAttention_35021163331889
// MI455X (gfx1250) — matrix operand order repaired and checked by reading, not compiled, not run on hardware
//
#include <hip/hip_runtime.h>

typedef __attribute__((ext_vector_type(16))) _Float16 v16h;
typedef __attribute__((ext_vector_type(8)))  float    v8f;

#define NEG_INF (-1e30f)

#define BB 64
#define LCC 1024
#define LQQ 128
#define DDD 256

// LDS layout (element counts). Rows padded by 8 halves so that per-row bank
// advance is +4 banks -> half-wave b128 B-frag loads are conflict-minimal.
#define QRM_STRIDE 264                    // Q row-major: 256 + 8
#define QT_STRIDE  136                    // Q transposed: 128 + 8
#define P_STRIDE   136                    // probs tile: 128 + 8
#define QRM_ELEMS (LQQ * QRM_STRIDE)      // 33792 halves
#define QT_ELEMS  (DDD * QT_STRIDE)       // 34816 halves
#define P_ELEMS   (8 * 16 * P_STRIDE)     // 17408 halves
#define SMEM_BYTES (((QRM_ELEMS) + (QT_ELEMS) + (P_ELEMS)) * 2 + (512 + 128 + 128) * 4)

union BFrag { v16h v; uint4 q[2]; };
union AFrag { v16h v; _Float16 h[16]; };
union HPack { uint u; _Float16 h[2]; };
union H8    { uint4 q; _Float16 h[8]; };

__global__ __launch_bounds__(256, 1)
void cqattn_wmma(const float* __restrict__ Cg, const float* __restrict__ Qg,
                 const float* __restrict__ qmaskg, const float* __restrict__ wg,
                 float* __restrict__ outg)
{
    extern __shared__ char smem_raw[];
    _Float16* Qrm = reinterpret_cast<_Float16*>(smem_raw);   // Q row-major f16
    _Float16* Qt  = Qrm + QRM_ELEMS;                         // Q transposed f16
    _Float16* Pb  = Qt + QT_ELEMS;                           // per-wave prob tiles
    float* wl  = reinterpret_cast<float*>(Pb + P_ELEMS);     // [0:256)=w2 [256:512)=w3
    float* tb  = wl + 512;                                   // bias t_j = Q_j . w2
    float* qmL = tb + 128;                                   // qmask[b]

    const int tid   = threadIdx.x;
    const int b     = blockIdx.x >> 3;
    const int i0blk = (blockIdx.x & 7) * 128;

    const float* Qb = Qg + (size_t)b * LQQ * DDD;
    const float* Cb = Cg + (size_t)b * LCC * DDD;

    // ---- stage weights and mask ----
    wl[tid]       = wg[256 + tid];   // w2
    wl[256 + tid] = wg[512 + tid];   // w3
    if (tid < 128) qmL[tid] = qmaskg[b * LQQ + tid];

    // ---- stage Q[b] (128x256 f32) into Qrm (f16) and Qt (f16 transposed) ----
    {
        const float4* Q4 = reinterpret_cast<const float4*>(Qb);
        #pragma unroll
        for (int it = 0; it < 32; ++it) {
            const int idx4 = tid + it * 256;
            const float4 v = Q4[idx4];
            const int flat = idx4 << 2;
            const int j = flat >> 8;     // Q row
            const int d = flat & 255;    // Q col (multiple of 4)
            const _Float16 h0 = (_Float16)v.x, h1 = (_Float16)v.y,
                           h2 = (_Float16)v.z, h3 = (_Float16)v.w;
            HPack p0, p1;
            p0.h[0] = h0; p0.h[1] = h1;
            p1.h[0] = h2; p1.h[1] = h3;
            uint* qr = reinterpret_cast<uint*>(Qrm + j * QRM_STRIDE + d);
            qr[0] = p0.u; qr[1] = p1.u;
            Qt[(d + 0) * QT_STRIDE + j] = h0;
            Qt[(d + 1) * QT_STRIDE + j] = h1;
            Qt[(d + 2) * QT_STRIDE + j] = h2;
            Qt[(d + 3) * QT_STRIDE + j] = h3;
        }
    }
    __syncthreads();

    // ---- bias t_j = Q[j] . w2  (C@w1 term cancels in softmax: dropped) ----
    if (tid < 128) {
        const uint4* qr = reinterpret_cast<const uint4*>(Qrm + tid * QRM_STRIDE);
        float s = 0.f;
        #pragma unroll 4
        for (int it = 0; it < 32; ++it) {         // 8 halves / iter
            H8 u; u.q = qr[it];
            const float4 wa = *reinterpret_cast<const float4*>(wl + it * 8);
            const float4 wb = *reinterpret_cast<const float4*>(wl + it * 8 + 4);
            s += (float)u.h[0] * wa.x + (float)u.h[1] * wa.y +
                 (float)u.h[2] * wa.z + (float)u.h[3] * wa.w +
                 (float)u.h[4] * wb.x + (float)u.h[5] * wb.y +
                 (float)u.h[6] * wb.z + (float)u.h[7] * wb.w;
        }
        tb[tid] = s;
    }
    __syncthreads();

    const int wave = tid >> 5;
    const int lane = tid & 31;
    const int ln   = lane & 15;   // N index / A-row index
    const int hi   = lane >> 4;   // half-wave select
    const int i0   = i0blk + wave * 16;

    // ================= GEMM1: scores(16x128) = (C tile ⊙ w3) @ Q^T =================
    v8f acc[8];
    #pragma unroll
    for (int nt = 0; nt < 8; ++nt)
        #pragma unroll
        for (int r = 0; r < 8; ++r) acc[nt][r] = 0.f;

    const float* crow = Cg + ((size_t)b * LCC + (i0 + ln)) * DDD;
    const _Float16* qbase = Qrm + ln * QRM_STRIDE + hi * 8;

    #pragma unroll
    for (int kk = 0; kk < 8; ++kk) {
        const int k0 = kk * 32;
        const int da = k0 + hi * 8;           // A layout: halves 0..7 = K da..da+7
        AFrag af;                             //           halves 8..15 = K da+16..da+23
        const float4 c0 = *reinterpret_cast<const float4*>(crow + da);
        const float4 c1 = *reinterpret_cast<const float4*>(crow + da + 4);
        const float4 c2 = *reinterpret_cast<const float4*>(crow + da + 16);
        const float4 c3 = *reinterpret_cast<const float4*>(crow + da + 20);
        const float4 w0 = *reinterpret_cast<const float4*>(wl + 256 + da);
        const float4 w1 = *reinterpret_cast<const float4*>(wl + 256 + da + 4);
        const float4 w2v = *reinterpret_cast<const float4*>(wl + 256 + da + 16);
        const float4 w3v = *reinterpret_cast<const float4*>(wl + 256 + da + 20);
        af.h[0]  = (_Float16)(c0.x * w0.x);  af.h[1]  = (_Float16)(c0.y * w0.y);
        af.h[2]  = (_Float16)(c0.z * w0.z);  af.h[3]  = (_Float16)(c0.w * w0.w);
        af.h[4]  = (_Float16)(c1.x * w1.x);  af.h[5]  = (_Float16)(c1.y * w1.y);
        af.h[6]  = (_Float16)(c1.z * w1.z);  af.h[7]  = (_Float16)(c1.w * w1.w);
        af.h[8]  = (_Float16)(c2.x * w2v.x); af.h[9]  = (_Float16)(c2.y * w2v.y);
        af.h[10] = (_Float16)(c2.z * w2v.z); af.h[11] = (_Float16)(c2.w * w2v.w);
        af.h[12] = (_Float16)(c3.x * w3v.x); af.h[13] = (_Float16)(c3.y * w3v.y);
        af.h[14] = (_Float16)(c3.z * w3v.z); af.h[15] = (_Float16)(c3.w * w3v.w);

        // B layout: lane n holds column n (= Q row nt*16+n), K 8h..8h+7 and 16+8h..16+8h+7.
        // Double-buffered so the next frag's ds_loads issue before each wmma.
        BFrag bf[2];
        {
            const _Float16* qp = qbase + k0;           // nt = 0
            bf[0].q[0] = *reinterpret_cast<const uint4*>(qp);
            bf[0].q[1] = *reinterpret_cast<const uint4*>(qp + 16);
        }
        #pragma unroll
        for (int nt = 0; nt < 8; ++nt) {
            if (nt < 7) {
                const _Float16* qp = qbase + (nt + 1) * 16 * QRM_STRIDE + k0;
                bf[(nt + 1) & 1].q[0] = *reinterpret_cast<const uint4*>(qp);
                bf[(nt + 1) & 1].q[1] = *reinterpret_cast<const uint4*>(qp + 16);
            }
            acc[nt] = __builtin_amdgcn_wmma_f32_16x16x32_f16(
                false, af.v, false, bf[nt & 1].v, (short)0, acc[nt], false, false);
        }
    }

    // ---- bias + mask:  s = (s + t_j)*(1-m) + m*NEG_INF  (matches reference) ----
    #pragma unroll
    for (int nt = 0; nt < 8; ++nt) {
        const float t    = tb[nt * 16 + ln];
        const float m    = qmL[nt * 16 + ln];
        const float keep = 1.f - m;
        const float madd = m * NEG_INF;
        #pragma unroll
        for (int r = 0; r < 8; ++r) acc[nt][r] = (acc[nt][r] + t) * keep + madd;
    }

    // ---- softmax over j (row = (r, half); 16 lanes per half hold the N values) ----
    float rmax[8], rsum[8], rinv[8];
    #pragma unroll
    for (int r = 0; r < 8; ++r) {
        float m = acc[0][r];
        #pragma unroll
        for (int nt = 1; nt < 8; ++nt) m = fmaxf(m, acc[nt][r]);
        rmax[r] = m;
    }
    #pragma unroll
    for (int off = 8; off >= 1; off >>= 1)
        #pragma unroll
        for (int r = 0; r < 8; ++r)
            rmax[r] = fmaxf(rmax[r], __shfl_xor(rmax[r], off, 32));
    #pragma unroll
    for (int r = 0; r < 8; ++r) rsum[r] = 0.f;
    #pragma unroll
    for (int nt = 0; nt < 8; ++nt)
        #pragma unroll
        for (int r = 0; r < 8; ++r) {
            const float e = __expf(acc[nt][r] - rmax[r]);
            acc[nt][r] = e;
            rsum[r] += e;
        }
    #pragma unroll
    for (int off = 8; off >= 1; off >>= 1)
        #pragma unroll
        for (int r = 0; r < 8; ++r) rsum[r] += __shfl_xor(rsum[r], off, 32);
    #pragma unroll
    for (int r = 0; r < 8; ++r) rinv[r] = 1.f / rsum[r];

    // ---- stage probs (unnormalized exp) as f16 A-matrix source ----
    _Float16* Pw = Pb + wave * 16 * P_STRIDE;
    #pragma unroll
    for (int nt = 0; nt < 8; ++nt)
        #pragma unroll
        for (int r = 0; r < 8; ++r)
            Pw[(r + 8 * hi) * P_STRIDE + nt * 16 + ln] = (_Float16)acc[nt][r];
    __syncthreads();

    // ================= GEMM2: A(16x256) = P(16x128) @ Q(128x256) =================
    // A-frags (probabilities) are invariant across all 16 dt tiles: hoist.
    BFrag pf[4];
    #pragma unroll
    for (int kj = 0; kj < 4; ++kj) {
        const _Float16* pp = Pw + ln * P_STRIDE + kj * 32 + hi * 8;
        pf[kj].q[0] = *reinterpret_cast<const uint4*>(pp);
        pf[kj].q[1] = *reinterpret_cast<const uint4*>(pp + 16);
    }

    #pragma unroll 2
    for (int dt = 0; dt < 16; ++dt) {
        // batch all 4 B-frag loads so dscnt waits stagger instead of hitting 0
        BFrag bf[4];
        const _Float16* qtb = Qt + (dt * 16 + ln) * QT_STRIDE + hi * 8;
        #pragma unroll
        for (int kj = 0; kj < 4; ++kj) {
            bf[kj].q[0] = *reinterpret_cast<const uint4*>(qtb + kj * 32);
            bf[kj].q[1] = *reinterpret_cast<const uint4*>(qtb + kj * 32 + 16);
        }
        v8f f;
        #pragma unroll
        for (int r = 0; r < 8; ++r) f[r] = 0.f;
        #pragma unroll
        for (int kj = 0; kj < 4; ++kj)
            f = __builtin_amdgcn_wmma_f32_16x16x32_f16(
                false, pf[kj].v, false, bf[kj].v, (short)0, f, false, false);

        const int col = dt * 16 + ln;
        #pragma unroll
        for (int r = 0; r < 8; ++r) {
            const int row = i0 + r + 8 * hi;
            const float a  = f[r] * rinv[r];
            const float cv = Cb[(size_t)row * DDD + col];
            float* op = outg + ((size_t)(b * LCC + row)) * (2 * DDD) + col;
            op[0]   = a;        // A
            op[DDD] = cv * a;   // C ⊙ A
        }
    }
}

extern "C" void kernel_launch(void* const* d_in, const int* in_sizes, int n_in,
                              void* d_out, int out_size, void* d_ws, size_t ws_size,
                              hipStream_t stream) {
    (void)in_sizes; (void)n_in; (void)out_size; (void)d_ws; (void)ws_size;
    const float* C     = (const float*)d_in[0];
    const float* Q     = (const float*)d_in[1];
    // d_in[2] = cmask: unused by the reference computation
    const float* qmask = (const float*)d_in[3];
    const float* w     = (const float*)d_in[4];
    float* out = (float*)d_out;

    dim3 grid(BB * (LCC / 128));   // 512 workgroups: (batch, 128-row C tile)
    dim3 block(256);               // 8 waves of 32
    cqattn_wmma<<<grid, block, SMEM_BYTES, stream>>>(C, Q, qmask, w, out);
}
